// RingEncoder_79585743994963
// MI455X (gfx1250) — matrix operand order repaired and checked by reading, not compiled, not run on hardware
//
#include <hip/hip_runtime.h>
#include <hip/hip_bf16.h>

// ---------------------------------------------------------------------------
// RingEncoder for MI455X (gfx1250, wave32, WMMA).
//  - conv1x1 GEMMs: v_wmma_f32_16x16x32_bf16, f32 accumulate.
//  - activations kept channel-contiguous (B,N,Cpad) bf16, zero-padded to
//    multiples of 32 -> WMMA B-fragment is ONE contiguous 32B load per lane.
//  - weights pre-converted to bf16 (Co,Cpad) -> A-fragment is two 16B loads.
//  - BN folded into streaming convert passes after fused stats GEMMs.
//  - (B,1024,N) activations never materialized: stats pass + fused BN/max
//    pass recompute the GEMM (saves ~1.6 GB HBM traffic).
// ---------------------------------------------------------------------------

#define B_ 8
#define N_ 8192
#define C_ 5
#define NR_ 16
#define MR_ 1520
#define S_ 512            // N_/NR_
#define L_ (NR_ * MR_)    // 24320
#define EPS_ 1e-5f
#define BN_ (B_ * N_)     // 65536 columns

typedef __attribute__((ext_vector_type(16))) __bf16 v16bf;
typedef __attribute__((ext_vector_type(8)))  __bf16 v8bf;
typedef __attribute__((ext_vector_type(8)))  float  v8f;

#define MODE_STORE      0
#define MODE_STATS_ONLY 1
#define MODE_MAX        2

// order-preserving float -> uint key so atomicMax(unsigned) == float max
__device__ __forceinline__ unsigned fkey(float f) {
  unsigned u = __float_as_uint(f);
  return (u & 0x80000000u) ? ~u : (u | 0x80000000u);
}
__device__ __forceinline__ float kdec(unsigned k) {
  return (k & 0x80000000u) ? __uint_as_float(k & 0x7FFFFFFFu)
                           : __uint_as_float(~k);
}

// ---------------------------------------------------------------------------
// WMMA GEMM: one wave computes a 64(co) x 16(n) tile, K in steps of 32.
//   X: bf16 (B, N, Cpad) channel-contiguous, zero padded
//   W: bf16 (Co, Cpad) (or (B, Co, Cpad) if perBatch)
// Modes: STORE (f32 (B,N,Co) and/or bf16 (B,N,Co)), STATS_ONLY, MAX.
// ---------------------------------------------------------------------------
__global__ __launch_bounds__(256) void gemm_wmma_kernel(
    const __bf16* __restrict__ Xbf,
    const __bf16* __restrict__ Wbf,
    const float*  __restrict__ bias,     // (Co) or nullptr
    float*  __restrict__ Yf,             // (B,N,Co) f32 or nullptr
    __bf16* __restrict__ Ybf,            // (B,N,Co) bf16 or nullptr
    float*  __restrict__ statAcc,        // (2*Co) sum|sumsq or nullptr
    const float* __restrict__ outScSh,   // (2*Co) for MODE_MAX
    int outRelu,
    unsigned* __restrict__ maxKeys,      // (B*Co) for MODE_MAX
    int Cpad, int Co, int perBatch, int mode)
{
  const int lane = threadIdx.x & 31;
  const int wave = blockIdx.x * (blockDim.x >> 5) + (threadIdx.x >> 5);
  const int coGroups = Co >> 6;              // 64 co rows per wave
  const int colTiles = BN_ >> 4;
  if (wave >= coGroups * colTiles) return;   // whole-wave uniform exit
  const int tcg  = wave % coGroups;
  const int tcol = wave / coGroups;
  const int co0  = tcg << 6;
  const long col0 = (long)tcol << 4;
  const int b  = (int)(col0 / N_);
  const int n0 = (int)(col0 % N_);
  const int half = lane >> 4;
  const int rl   = lane & 15;
  const int n    = n0 + rl;

  const __bf16* Wb   = Wbf + (perBatch ? (size_t)b * Co * Cpad : 0);
  const __bf16* Wrow = Wb + ((size_t)co0 + rl) * Cpad;   // A row for this lane
  const size_t  jStr = (size_t)16 * Cpad;
  const __bf16* Xrow = Xbf + ((size_t)b * N_ + n) * Cpad;

  v8f acc[4];
#pragma unroll
  for (int j = 0; j < 4; ++j)
    acc[j] = (v8f){0.f, 0.f, 0.f, 0.f, 0.f, 0.f, 0.f, 0.f};

  const int ksteps = Cpad >> 5;
  for (int ks = 0; ks < ksteps; ++ks) {
    const int k0 = ks << 5;
    // B fragment (32x16 KxN): lane col = rl, K = half*16 .. half*16+15
    v16bf bv = *(const v16bf*)(Xrow + k0 + half * 16);
    if (ks + 1 < ksteps)
      __builtin_prefetch(Xrow + k0 + 32, 0, 0);
#pragma unroll
    for (int j = 0; j < 4; ++j) {
      // A fragment (16x32 MxK): two contiguous 8-elt runs per lane
      const __bf16* Ar = Wrow + j * jStr + k0;
      v8bf lo = *(const v8bf*)(Ar + half * 16);
      v8bf hi = *(const v8bf*)(Ar + half * 16 + 8);
      v16bf av;
#pragma unroll
      for (int e = 0; e < 8; ++e) { av[e] = lo[e]; av[8 + e] = hi[e]; }
      acc[j] = __builtin_amdgcn_wmma_f32_16x16x32_bf16(
          false, av, false, bv, (short)0, acc[j], false, false);
    }
  }

  // D layout: VGPR v holds row M = half*8 + v (within each 16-row j tile)
#pragma unroll
  for (int j = 0; j < 4; ++j) {
    const int coB = co0 + j * 16 + half * 8;   // 8 consecutive channels
    float vals[8];
#pragma unroll
    for (int v = 0; v < 8; ++v)
      vals[v] = acc[j][v] + (bias ? bias[coB + v] : 0.f);
    if (mode == MODE_STORE) {
      if (Yf) {
        float* yp = Yf + ((size_t)b * N_ + n) * Co + coB;
#pragma unroll
        for (int v = 0; v < 8; ++v) yp[v] = vals[v];
      }
      if (Ybf) {
        __bf16* yb = Ybf + ((size_t)b * N_ + n) * Co + coB;
#pragma unroll
        for (int v = 0; v < 8; ++v) yb[v] = (__bf16)vals[v];
      }
    }
    if (statAcc) {                 // reduce 16 columns, 1 atomic pair per channel
#pragma unroll
      for (int v = 0; v < 8; ++v) {
        float s = vals[v], sq = vals[v] * vals[v];
#pragma unroll
        for (int m = 1; m < 16; m <<= 1) {
          s  += __shfl_xor(s,  m, 32);
          sq += __shfl_xor(sq, m, 32);
        }
        if (rl == 0) {
          atomicAdd(&statAcc[coB + v], s);
          atomicAdd(&statAcc[Co + coB + v], sq);
        }
      }
    }
    if (mode == MODE_MAX) {
#pragma unroll
      for (int v = 0; v < 8; ++v) {
        float t = vals[v] * outScSh[coB + v] + outScSh[Co + coB + v];
        if (outRelu) t = fmaxf(t, 0.f);
#pragma unroll
        for (int m = 1; m < 16; m <<= 1)
          t = fmaxf(t, __shfl_xor(t, m, 32));
        if (rl == 0)
          atomicMax(&maxKeys[(size_t)b * Co + coB + v], fkey(t));
      }
    }
  }
}

// ---------------------------------------------------------------------------
__global__ void fill_u32_kernel(unsigned* __restrict__ p, unsigned v, long n) {
  long t = (long)blockIdx.x * blockDim.x + threadIdx.x;
  if (t < n) p[t] = v;
}

__global__ void bn_finalize_kernel(const float* __restrict__ acc,
                                   const float* __restrict__ g,
                                   const float* __restrict__ bta,
                                   float* __restrict__ scsh,
                                   int Co, float invCount) {
  int c = blockIdx.x * blockDim.x + threadIdx.x;
  if (c >= Co) return;
  float mean = acc[c] * invCount;
  float var  = acc[Co + c] * invCount - mean * mean;
  float sc   = g[c] * rsqrtf(var + EPS_);
  scsh[c]      = sc;
  scsh[Co + c] = bta[c] - mean * sc;
}

// streaming BN(+relu) apply + f32 -> bf16 (with zero channel padding)
__global__ void bn_convert_kernel(const float* __restrict__ in,  // (B,N,Co)
                                  const float* __restrict__ scsh,
                                  int relu,
                                  __bf16* __restrict__ out,      // (B,N,Cpad)
                                  int Co, int Cpad, long total) {
  long t = (long)blockIdx.x * blockDim.x + threadIdx.x;
  if (t >= total) return;
  int c = (int)(t % Cpad);
  long bn = t / Cpad;
  float v = 0.f;
  if (c < Co) {
    v = in[bn * Co + c] * scsh[c] + scsh[Co + c];
    if (relu) v = fmaxf(v, 0.f);
  }
  out[t] = (__bf16)v;
}

// x (B,C,N) f32 -> (B,N,32) bf16 zero-padded
__global__ void x_to_bf_kernel(const float* __restrict__ x,
                               __bf16* __restrict__ out) {
  long t = (long)blockIdx.x * blockDim.x + threadIdx.x;
  if (t >= (long)BN_ * 32) return;
  int c = (int)(t & 31);
  long bn = t >> 5;
  int n = (int)(bn % N_), b = (int)(bn / N_);
  out[t] = (__bf16)(c < C_ ? x[((size_t)b * C_ + c) * N_ + n] : 0.f);
}

// xt = concat(xyz @ trans, feat) -> (B,N,32) bf16 zero-padded
__global__ void transform_pts_bf_kernel(const float* __restrict__ x,
                                        const float* __restrict__ trans, // (B,3,3)
                                        __bf16* __restrict__ out) {
  long t = (long)blockIdx.x * blockDim.x + threadIdx.x;
  if (t >= (long)BN_) return;
  int n = (int)(t % N_), b = (int)(t / N_);
  const float* xb = x + (size_t)b * C_ * N_;
  const float* tr = trans + b * 9;
  float p0 = xb[0 * N_ + n], p1 = xb[1 * N_ + n], p2 = xb[2 * N_ + n];
  __bf16* ob = out + t * 32;
#pragma unroll
  for (int e = 0; e < 3; ++e)
    ob[e] = (__bf16)(p0 * tr[e] + p1 * tr[3 + e] + p2 * tr[6 + e]);
  ob[3] = (__bf16)xb[3 * N_ + n];
  ob[4] = (__bf16)xb[4 * N_ + n];
#pragma unroll
  for (int e = 5; e < 32; ++e) ob[e] = (__bf16)0.f;
}

// weights f32 (Co,Cin) -> bf16 (Co,Cpad) zero-padded
__global__ void wcvt_kernel(const float* __restrict__ W, __bf16* __restrict__ out,
                            int Cin, int Cpad, long total) {
  long t = (long)blockIdx.x * blockDim.x + threadIdx.x;
  if (t >= total) return;
  int c = (int)(t % Cpad);
  long co = t / Cpad;
  out[t] = (__bf16)(c < Cin ? W[co * Cin + c] : 0.f);
}

// trans_feat (B,64,64)[c,d] -> per-batch weight bf16 (B, Co=d, Cin=c)
__global__ void tf_transpose_kernel(const float* __restrict__ tf,
                                    __bf16* __restrict__ out) {
  long t = (long)blockIdx.x * blockDim.x + threadIdx.x;
  if (t >= (long)B_ * 64 * 64) return;
  int c = (int)(t & 63);
  long rest = t >> 6;
  int d = (int)(rest & 63);
  int b = (int)(rest >> 6);
  out[t] = (__bf16)tf[(size_t)b * 4096 + c * 64 + d];
}

// STN tail: 3 FCs with batch(=8)-dim BN, single workgroup, all in LDS.
__global__ __launch_bounds__(512) void stn_fc_kernel(
    const unsigned* __restrict__ mkeys,  // (8*1024) max keys
    const float* __restrict__ f1w, const float* __restrict__ f1b,
    const float* __restrict__ f2w, const float* __restrict__ f2b,
    const float* __restrict__ f3w, const float* __restrict__ f3b,
    const float* __restrict__ bn4g, const float* __restrict__ bn4b,
    const float* __restrict__ bn5g, const float* __restrict__ bn5b,
    float* __restrict__ outp, int k)
{
  __shared__ float s_in[8 * 1024];
  __shared__ float s_h1[8 * 512];
  __shared__ float s_h2[8 * 256];
  const int tid = threadIdx.x;
  for (int i = tid; i < 8 * 1024; i += 512) s_in[i] = kdec(mkeys[i]);
  __syncthreads();
  for (int t = tid; t < 8 * 512; t += 512) {
    int b = t >> 9, o = t & 511;
    float a = f1b[o];
    const float* wr = f1w + (size_t)o * 1024;
    const float* xr = s_in + b * 1024;
    for (int c = 0; c < 1024; ++c) a = fmaf(wr[c], xr[c], a);
    s_h1[t] = a;
  }
  __syncthreads();
  if (tid < 512) {
    float mu = 0.f;
    for (int b = 0; b < 8; ++b) mu += s_h1[b * 512 + tid];
    mu *= 0.125f;
    float var = 0.f;
    for (int b = 0; b < 8; ++b) { float d = s_h1[b * 512 + tid] - mu; var += d * d; }
    var *= 0.125f;
    float sc = bn4g[tid] * rsqrtf(var + EPS_);
    float sh = bn4b[tid] - mu * sc;
    for (int b = 0; b < 8; ++b)
      s_h1[b * 512 + tid] = fmaxf(s_h1[b * 512 + tid] * sc + sh, 0.f);
  }
  __syncthreads();
  for (int t = tid; t < 8 * 256; t += 512) {
    int b = t >> 8, o = t & 255;
    float a = f2b[o];
    const float* wr = f2w + (size_t)o * 512;
    const float* xr = s_h1 + b * 512;
    for (int c = 0; c < 512; ++c) a = fmaf(wr[c], xr[c], a);
    s_h2[t] = a;
  }
  __syncthreads();
  if (tid < 256) {
    float mu = 0.f;
    for (int b = 0; b < 8; ++b) mu += s_h2[b * 256 + tid];
    mu *= 0.125f;
    float var = 0.f;
    for (int b = 0; b < 8; ++b) { float d = s_h2[b * 256 + tid] - mu; var += d * d; }
    var *= 0.125f;
    float sc = bn5g[tid] * rsqrtf(var + EPS_);
    float sh = bn5b[tid] - mu * sc;
    for (int b = 0; b < 8; ++b)
      s_h2[b * 256 + tid] = fmaxf(s_h2[b * 256 + tid] * sc + sh, 0.f);
  }
  __syncthreads();
  const int kk = k * k;
  for (int t = tid; t < 8 * kk; t += 512) {
    int b = t / kk, i = t % kk;
    float a = f3b[i];
    const float* wr = f3w + (size_t)i * 256;
    const float* xr = s_h2 + b * 256;
    for (int c = 0; c < 256; ++c) a = fmaf(wr[c], xr[c], a);
    if (i % (k + 1) == 0) a += 1.f;      // + eye(k)
    outp[(size_t)b * kk + i] = a;
  }
}

// xu_flat[b][(r*1520+i)*64+c] = h2f[b][r*512+idx][c]   (h2f is (B,N,64))
__global__ void gather_kernel(const float* __restrict__ h2f,
                              const int* __restrict__ idx,
                              float* __restrict__ xu) {
  long t = (long)blockIdx.x * blockDim.x + threadIdx.x;
  if (t >= (long)B_ * NR_ * MR_ * 64) return;
  int cc = (int)(t & 63);
  long rest = t >> 6;
  int ii = (int)(rest % MR_); rest /= MR_;
  int rr = (int)(rest % NR_);
  int b  = (int)(rest / NR_);
  int id = idx[(b * NR_ + rr) * MR_ + ii];
  xu[t] = h2f[((size_t)b * N_ + rr * S_ + id) * 64 + cc];
}

// grouped ring conv stats: r[b,ch,l] = bias + sum_c4 w[g,o,c4]*xu[b,(g*4+c4)*L+l]
__global__ __launch_bounds__(256) void ring_stats_kernel(
    const float* __restrict__ xu, const float* __restrict__ rcw,
    const float* __restrict__ rcb, float* __restrict__ stats) {
  const int lane = threadIdx.x & 31;
  const int wave = blockIdx.x * (blockDim.x >> 5) + (threadIdx.x >> 5);
  const int nChunks = L_ / 256;                 // 95
  if (wave >= B_ * 128 * nChunks) return;
  int lc = wave % nChunks;
  int rest = wave / nChunks;
  int ch = rest % 128, b = rest / 128;
  int g = ch >> 3;
  const float* wv = &rcw[ch * 4];
  float bias = rcb[ch];
  size_t base = (size_t)b * 64 * L_;
  float s = 0.f, sq = 0.f;
  for (int c = 0; c < 8; ++c) {
    int l = lc * 256 + c * 32 + lane;
    float a = bias;
#pragma unroll
    for (int c4 = 0; c4 < 4; ++c4)
      a = fmaf(wv[c4], xu[base + (size_t)(g * 4 + c4) * L_ + l], a);
    s += a; sq += a * a;
  }
#pragma unroll
  for (int m = 1; m < 32; m <<= 1) {
    s += __shfl_xor(s, m, 32); sq += __shfl_xor(sq, m, 32);
  }
  if (lane == 0) { atomicAdd(&stats[ch], s); atomicAdd(&stats[128 + ch], sq); }
}

// rm[b,ch,ring] = max_i BN(r[b,ch,ring*1520+i])  (r recomputed, never stored)
__global__ __launch_bounds__(256) void ring_max_kernel(
    const float* __restrict__ xu, const float* __restrict__ rcw,
    const float* __restrict__ rcb, const float* __restrict__ scsh,
    float* __restrict__ rm) {
  __shared__ float red[256];
  int id = blockIdx.x;
  int ring = id % NR_; int rest = id / NR_;
  int ch = rest % 128, b = rest / 128;
  int g = ch >> 3;
  const float* wv = &rcw[ch * 4];
  float bias = rcb[ch];
  float sc = scsh[ch], sh = scsh[128 + ch];
  size_t base = (size_t)b * 64 * L_;
  float mx = -3.4e38f;
  for (int ii = threadIdx.x; ii < MR_; ii += 256) {
    int l = ring * MR_ + ii;
    float a = bias;
#pragma unroll
    for (int c4 = 0; c4 < 4; ++c4)
      a = fmaf(wv[c4], xu[base + (size_t)(g * 4 + c4) * L_ + l], a);
    mx = fmaxf(mx, a * sc + sh);
  }
  red[threadIdx.x] = mx;
  __syncthreads();
  for (int s2 = 128; s2 > 0; s2 >>= 1) {
    if ((int)threadIdx.x < s2)
      red[threadIdx.x] = fmaxf(red[threadIdx.x], red[threadIdx.x + s2]);
    __syncthreads();
  }
  if (threadIdx.x == 0) rm[((size_t)b * 128 + ch) * NR_ + ring] = red[0];
}

// out(B,1216,N): [0,1024)=x1 broadcast, [1024,1152)=x2 ring max, [1152,1216)=h
__global__ void assemble_kernel(const unsigned* __restrict__ gKeys,
                                const float* __restrict__ rm,
                                const float* __restrict__ h2f,   // (B,N,64)
                                float* __restrict__ out) {
  long t = (long)blockIdx.x * blockDim.x + threadIdx.x;
  if (t >= (long)B_ * 1216 * N_) return;
  int n = (int)(t % N_);
  long rest = t / N_;
  int ch = (int)(rest % 1216), b = (int)(rest / 1216);
  float v;
  if (ch < 1024) {
    v = kdec(gKeys[b * 1024 + ch]);
  } else if (ch < 1152) {
    v = rm[((size_t)b * 128 + (ch - 1024)) * NR_ + (n / S_)];
  } else {
    v = h2f[((size_t)b * N_ + n) * 64 + (ch - 1152)];
  }
  out[t] = v;
}

// ---------------------------------------------------------------------------
extern "C" void kernel_launch(void* const* d_in, const int* in_sizes, int n_in,
                              void* d_out, int out_size, void* d_ws, size_t ws_size,
                              hipStream_t stream) {
  (void)in_sizes; (void)n_in; (void)out_size; (void)ws_size;
  // Inputs flattened in setup_inputs() dict order (params depth-first):
  // 0:x 1:ring(unused) 2:upsample_idx, 3..24 stn leaves, 25..46 fstn leaves,
  // 47..62 top-level params.
  auto F = [&](int i) { return (const float*)d_in[i]; };
  const float* x   = F(0);
  const int*   uix = (const int*)d_in[2];
  const int S3 = 3, FS = 25;
  const float *conv1w = F(47), *conv1b = F(48), *bn1g = F(49), *bn1b = F(50);
  const float *gc1w = F(51), *gc1b = F(52), *gbn1g = F(53), *gbn1b = F(54);
  const float *gc2w = F(55), *gc2b = F(56), *gbn2g = F(57), *gbn2b = F(58);
  const float *rcw = F(59), *rcb = F(60), *rbng = F(61), *rbnb = F(62);

  // ---- workspace carve (256B aligned blocks) ----
  char* basep = (char*)d_ws;
  size_t ob = 0;
  auto carve = [&](size_t bytes) {
    void* p = basep + ob;
    ob += (bytes + 255) & ~(size_t)255;
    return p;
  };
  float*  U64A   = (float*)carve((size_t)BN_ * 64 * 4);    // raw f32 (B,N,64)
  float*  Y1     = (float*)carve((size_t)BN_ * 64 * 4);
  float*  H2F    = (float*)carve((size_t)BN_ * 64 * 4);
  float*  YMID   = (float*)carve((size_t)BN_ * 128 * 4);
  float*  XU     = (float*)carve((size_t)B_ * 64 * L_ * 4);
  __bf16* XB     = (__bf16*)carve((size_t)BN_ * 32 * 2);
  __bf16* XTB    = (__bf16*)carve((size_t)BN_ * 32 * 2);
  __bf16* A64BF  = (__bf16*)carve((size_t)BN_ * 64 * 2);
  __bf16* Y1BF   = (__bf16*)carve((size_t)BN_ * 64 * 2);
  __bf16* H2BF   = (__bf16*)carve((size_t)BN_ * 64 * 2);
  __bf16* YMIDBF = (__bf16*)carve((size_t)BN_ * 128 * 2);
  __bf16* WS3C1  = (__bf16*)carve(64 * 32 * 2);
  __bf16* WS3C2  = (__bf16*)carve(128 * 64 * 2);
  __bf16* WS3C3  = (__bf16*)carve(1024 * 128 * 2);
  __bf16* WFC1   = (__bf16*)carve(64 * 64 * 2);
  __bf16* WFC2   = (__bf16*)carve(128 * 64 * 2);
  __bf16* WFC3   = (__bf16*)carve(1024 * 128 * 2);
  __bf16* WC1    = (__bf16*)carve(64 * 32 * 2);
  __bf16* WG1    = (__bf16*)carve(128 * 64 * 2);
  __bf16* WG2    = (__bf16*)carve(1024 * 128 * 2);
  __bf16* TFB    = (__bf16*)carve((size_t)B_ * 64 * 64 * 2);
  float*  STAT   = (float*)carve(2048 * 4);
  float*  S3B1   = (float*)carve(128 * 4);
  float*  S3B2   = (float*)carve(256 * 4);
  float*  S3B3   = (float*)carve(2048 * 4);
  float*  BN1    = (float*)carve(128 * 4);
  float*  FB1    = (float*)carve(128 * 4);
  float*  FB2    = (float*)carve(256 * 4);
  float*  FB3    = (float*)carve(2048 * 4);
  float*  GB1    = (float*)carve(256 * 4);
  float*  GB2    = (float*)carve(2048 * 4);
  float*  RBN    = (float*)carve(256 * 4);
  float*  RM     = (float*)carve((size_t)B_ * 128 * NR_ * 4);
  unsigned* MK   = (unsigned*)carve((size_t)B_ * 1024 * 4);
  unsigned* GK   = (unsigned*)carve((size_t)B_ * 1024 * 4);

  float* out    = (float*)d_out;
  float* transO = out + (size_t)B_ * 1216 * N_;   // (B,3,3)
  float* tfeatO = transO + (size_t)B_ * 9;        // (B,64,64)

  const float invBN = 1.f / (float)BN_;
  const float invBL = 1.f / (float)(B_ * L_);

  auto fill = [&](void* p, unsigned v, long n) {
    fill_u32_kernel<<<(unsigned)((n + 255) / 256), 256, 0, stream>>>((unsigned*)p, v, n);
  };
  auto gemm = [&](const __bf16* Xp, const __bf16* Wp, const float* bp,
                  float* Yfp, __bf16* Ybp, float* sAcc,
                  const float* oss, int oRelu, unsigned* mk,
                  int Cpad, int Co, int perB, int mode) {
    long waves = (long)(Co >> 6) * (long)(BN_ >> 4);
    gemm_wmma_kernel<<<(unsigned)((waves + 7) / 8), 256, 0, stream>>>(
        Xp, Wp, bp, Yfp, Ybp, sAcc, oss, oRelu, mk, Cpad, Co, perB, mode);
  };
  auto bnfin = [&](const float* gp, const float* bp, float* ss, int Co, float inv) {
    bn_finalize_kernel<<<(Co + 255) / 256, 256, 0, stream>>>(STAT, gp, bp, ss, Co, inv);
  };
  auto bncvt = [&](const float* in, const float* ss, int relu, __bf16* o2,
                   int Co, int Cpad) {
    long tot = (long)BN_ * Cpad;
    bn_convert_kernel<<<(unsigned)((tot + 255) / 256), 256, 0, stream>>>(
        in, ss, relu, o2, Co, Cpad, tot);
  };
  auto wcvt = [&](const float* Wp, __bf16* o2, int Cin, int Cpad, long Co) {
    long tot = Co * Cpad;
    wcvt_kernel<<<(unsigned)((tot + 255) / 256), 256, 0, stream>>>(Wp, o2, Cin, Cpad, tot);
  };

  // ---- one-time weight conversions (f32 -> padded bf16) ----
  wcvt(F(S3+0), WS3C1, C_, 32, 64);
  wcvt(F(S3+2), WS3C2, 64, 64, 128);
  wcvt(F(S3+4), WS3C3, 128, 128, 1024);
  wcvt(F(FS+0), WFC1, 64, 64, 64);
  wcvt(F(FS+2), WFC2, 64, 64, 128);
  wcvt(F(FS+4), WFC3, 128, 128, 1024);
  wcvt(conv1w, WC1, C_, 32, 64);
  wcvt(gc1w, WG1, 64, 64, 128);
  wcvt(gc2w, WG2, 128, 128, 1024);

  // ===== STN #1 on raw x (C=5 -> trans 3x3) =====
  {
    long tot = (long)BN_ * 32;
    x_to_bf_kernel<<<(unsigned)((tot + 255) / 256), 256, 0, stream>>>(x, XB);
  }
  fill(STAT, 0u, 128);
  gemm(XB, WS3C1, F(S3+1), U64A, nullptr, STAT, nullptr, 0, nullptr, 32, 64, 0, MODE_STORE);
  bnfin(F(S3+12), F(S3+13), S3B1, 64, invBN);
  bncvt(U64A, S3B1, 1, A64BF, 64, 64);
  fill(STAT, 0u, 256);
  gemm(A64BF, WS3C2, F(S3+3), YMID, nullptr, STAT, nullptr, 0, nullptr, 64, 128, 0, MODE_STORE);
  bnfin(F(S3+14), F(S3+15), S3B2, 128, invBN);
  bncvt(YMID, S3B2, 1, YMIDBF, 128, 128);
  fill(STAT, 0u, 2048);
  gemm(YMIDBF, WS3C3, F(S3+5), nullptr, nullptr, STAT, nullptr, 0, nullptr, 128, 1024, 0, MODE_STATS_ONLY);
  bnfin(F(S3+16), F(S3+17), S3B3, 1024, invBN);
  fill(MK, 0u, B_ * 1024);
  gemm(YMIDBF, WS3C3, F(S3+5), nullptr, nullptr, nullptr, S3B3, 1, MK, 128, 1024, 0, MODE_MAX);
  stn_fc_kernel<<<1, 512, 0, stream>>>(MK, F(S3+6), F(S3+7), F(S3+8), F(S3+9),
      F(S3+10), F(S3+11), F(S3+18), F(S3+19), F(S3+20), F(S3+21), transO, 3);

  // ===== transform points, conv1 =====
  transform_pts_bf_kernel<<<BN_ / 256, 256, 0, stream>>>(x, transO, XTB);
  fill(STAT, 0u, 128);
  gemm(XTB, WC1, conv1b, Y1, nullptr, STAT, nullptr, 0, nullptr, 32, 64, 0, MODE_STORE);
  bnfin(bn1g, bn1b, BN1, 64, invBN);
  bncvt(Y1, BN1, 1, Y1BF, 64, 64);

  // ===== STN #2 (feature STN) on Y1BF =====
  fill(STAT, 0u, 128);
  gemm(Y1BF, WFC1, F(FS+1), U64A, nullptr, STAT, nullptr, 0, nullptr, 64, 64, 0, MODE_STORE);
  bnfin(F(FS+12), F(FS+13), FB1, 64, invBN);
  bncvt(U64A, FB1, 1, A64BF, 64, 64);
  fill(STAT, 0u, 256);
  gemm(A64BF, WFC2, F(FS+3), YMID, nullptr, STAT, nullptr, 0, nullptr, 64, 128, 0, MODE_STORE);
  bnfin(F(FS+14), F(FS+15), FB2, 128, invBN);
  bncvt(YMID, FB2, 1, YMIDBF, 128, 128);
  fill(STAT, 0u, 2048);
  gemm(YMIDBF, WFC3, F(FS+5), nullptr, nullptr, STAT, nullptr, 0, nullptr, 128, 1024, 0, MODE_STATS_ONLY);
  bnfin(F(FS+16), F(FS+17), FB3, 1024, invBN);
  fill(MK, 0u, B_ * 1024);
  gemm(YMIDBF, WFC3, F(FS+5), nullptr, nullptr, nullptr, FB3, 1, MK, 128, 1024, 0, MODE_MAX);
  stn_fc_kernel<<<1, 512, 0, stream>>>(MK, F(FS+6), F(FS+7), F(FS+8), F(FS+9),
      F(FS+10), F(FS+11), F(FS+18), F(FS+19), F(FS+20), F(FS+21), tfeatO, 64);

  // ===== feature transform: H2[b,n,d] = sum_c Y1bn[b,n,c]*tf[b,c,d] =====
  {
    long tot = (long)B_ * 64 * 64;
    tf_transpose_kernel<<<(unsigned)((tot + 255) / 256), 256, 0, stream>>>(tfeatO, TFB);
  }
  gemm(Y1BF, TFB, nullptr, H2F, H2BF, nullptr, nullptr, 0, nullptr, 64, 64, 1, MODE_STORE);

  // ===== global branch =====
  fill(STAT, 0u, 256);
  gemm(H2BF, WG1, gc1b, YMID, nullptr, STAT, nullptr, 0, nullptr, 64, 128, 0, MODE_STORE);
  bnfin(gbn1g, gbn1b, GB1, 128, invBN);
  bncvt(YMID, GB1, 1, YMIDBF, 128, 128);
  fill(STAT, 0u, 2048);
  gemm(YMIDBF, WG2, gc2b, nullptr, nullptr, STAT, nullptr, 0, nullptr, 128, 1024, 0, MODE_STATS_ONLY);
  bnfin(gbn2g, gbn2b, GB2, 1024, invBN);
  fill(GK, 0u, B_ * 1024);
  gemm(YMIDBF, WG2, gc2b, nullptr, nullptr, nullptr, GB2, 0 /*no relu*/, GK, 128, 1024, 0, MODE_MAX);

  // ===== ring branch =====
  {
    long tot = (long)B_ * NR_ * MR_ * 64;
    gather_kernel<<<(unsigned)((tot + 255) / 256), 256, 0, stream>>>(H2F, uix, XU);
  }
  fill(STAT, 0u, 256);
  {
    long waves = (long)B_ * 128 * (L_ / 256);
    ring_stats_kernel<<<(unsigned)((waves + 7) / 8), 256, 0, stream>>>(XU, rcw, rcb, STAT);
  }
  bnfin(rbng, rbnb, RBN, 128, invBL);
  ring_max_kernel<<<B_ * 128 * NR_, 256, 0, stream>>>(XU, rcw, rcb, RBN, RM);

  // ===== final concat output =====
  {
    long tot = (long)B_ * 1216 * N_;
    assemble_kernel<<<(unsigned)((tot + 255) / 256), 256, 0, stream>>>(GK, RM, H2F, out);
  }
}
